// S4Layer_3753801417521
// MI455X (gfx1250) — matrix operand order repaired and checked by reading, not compiled, not run on hardware
//
#include <hip/hip_runtime.h>

#define BATCH   8
#define DMODEL  128
#define DSTATE  64
#define SEQLEN  2048
#define NCOL    (BATCH * SEQLEN)   // 16384 columns of the projection GEMM

typedef _Float16 v16h __attribute__((ext_vector_type(16)));
typedef _Float16 v8h  __attribute__((ext_vector_type(8)));
typedef float    v8f  __attribute__((ext_vector_type(8)));

// gfx1250 has a native V_TANH_F32 transcendental; use it if the builtin exists.
__device__ __forceinline__ float fast_tanh(float a) {
#if __has_builtin(__builtin_amdgcn_tanhf)
    return __builtin_amdgcn_tanhf(a);
#else
    float e = __expf(2.0f * a);           // v_exp_f32 path
    return 1.0f - 2.0f / (e + 1.0f);      // tanh(a) = 1 - 2/(e^{2a}+1)
#endif
}

// ---------------------------------------------------------------------------
// Phase 0: convert W (128x128 f32, row-major E x D) to f16 for WMMA.
// ---------------------------------------------------------------------------
__global__ void convert_w_kernel(const float* __restrict__ W,
                                 _Float16* __restrict__ Wh) {
    int i = blockIdx.x * blockDim.x + threadIdx.x;
    if (i < DMODEL * DMODEL) Wh[i] = (_Float16)W[i];
}

// ---------------------------------------------------------------------------
// Phase 1: recurrent scan. One wave32 per (b, d) pair; lane owns states
// n = lane and n = lane+32. h kept in registers (f32); per-step reduction
// over the 64 states via in-wave shfl_xor butterfly. Output written
// TRANSPOSED as f16: Yt[c][d], c = b*SEQLEN + l  (GEMM B-operand layout).
// ---------------------------------------------------------------------------
__global__ void s4_scan_kernel(const float* __restrict__ x,
                               const float* __restrict__ A,
                               const float* __restrict__ Bm,
                               const float* __restrict__ Cm,
                               const float* __restrict__ Dv,
                               _Float16* __restrict__ Yt) {
    const int wave = threadIdx.x >> 5;
    const int lane = threadIdx.x & 31;
    const int p    = blockIdx.x * (blockDim.x >> 5) + wave;  // 0..1023
    const int b    = p >> 7;        // batch
    const int d    = p & 127;       // model channel
    const int n0   = lane;
    const int n1   = lane + 32;

    // diag(A[d]) entries for this lane's two states
    const float a0 = A[(size_t)d * DSTATE * DSTATE + (size_t)n0 * DSTATE + n0];
    const float a1 = A[(size_t)d * DSTATE * DSTATE + (size_t)n1 * DSTATE + n1];
    const float b0 = Bm[d * DSTATE + n0];
    const float b1 = Bm[d * DSTATE + n1];
    const float c0 = Cm[d * DSTATE + n0];
    const float c1 = Cm[d * DSTATE + n1];
    const float dv = Dv[d];

    const float* __restrict__ xrow = x + ((size_t)b * DMODEL + d) * SEQLEN;
    _Float16* __restrict__ yout    = Yt + (size_t)b * SEQLEN * DMODEL + d;

    float h0 = 0.0f, h1 = 0.0f;
    for (int l = 0; l < SEQLEN; ++l) {
        float xt = xrow[l];                       // wave-uniform -> s_load
        h0 = fast_tanh(fmaf(a0, h0, b0 * xt));
        h1 = fast_tanh(fmaf(a1, h1, b1 * xt));
        float part = fmaf(c0, h0, c1 * h1);
        part += __shfl_xor(part, 16, 32);
        part += __shfl_xor(part,  8, 32);
        part += __shfl_xor(part,  4, 32);
        part += __shfl_xor(part,  2, 32);
        part += __shfl_xor(part,  1, 32);
        if (lane == 0)
            yout[(size_t)l * DMODEL] = (_Float16)(part + dv * xt);
    }
}

// ---------------------------------------------------------------------------
// Phase 2: projection GEMM  out[e, c] = sum_d W[e,d] * Yt[c,d]  (+ bias[e])
// via V_WMMA_F32_16X16X32_F16. One wave computes one 16(M=e) x 16(N=c) tile,
// accumulating K = d = 128 in 4 WMMA ops.
//
// A-operand (16-bit, 16x32 MxK), per ISA 7.12.2:
//   lane row M = lane&15; K chunks {0..7},{16..23} for lanes 0-15 and
//   {8..15},{24..31} for lanes 16-31  -> two contiguous v8h loads from Wh.
// B-operand (16-bit, 32x16 KxN):
//   lane col N = lane&15; K chunks {0..7},{16..23} (lanes 0-15) / {8..15},{24..31} (lanes 16-31)
//   -> two contiguous v8h loads from Yt[c][.].
// C/D f32 16x16: 8 VGPRs, N = lane&15, M = r + 8*(lane>=16).
// ---------------------------------------------------------------------------
__global__ void proj_wmma_kernel(const _Float16* __restrict__ Wh,
                                 const _Float16* __restrict__ Yt,
                                 const float* __restrict__ bias,
                                 float* __restrict__ out) {
    const int lane  = threadIdx.x & 31;
    const int wave  = threadIdx.x >> 5;
    const int ntile = blockIdx.x * (blockDim.x >> 5) + wave; // 0..1023
    const int mtile = blockIdx.y;                            // 0..7
    const int m0 = mtile * 16;
    const int c0 = ntile * 16;
    const int hi = lane >> 4;     // half-wave selector
    const int lm = lane & 15;

    v8f acc = {};
#pragma unroll
    for (int k0 = 0; k0 < DMODEL; k0 += 32) {
        union { v16h v; v8h h[2]; } a;
        const _Float16* ap = Wh + (size_t)(m0 + lm) * DMODEL + k0 + hi * 8;
        a.h[0] = *(const v8h*)(ap);        // K = k0 + hi*8 + {0..7}
        a.h[1] = *(const v8h*)(ap + 16);   // K = k0 + hi*8 + 16 + {0..7}

        union { v16h v; v8h h[2]; } bu;
        const _Float16* bp = Yt + (size_t)(c0 + lm) * DMODEL + k0 + hi * 8;
        bu.h[0] = *(const v8h*)(bp);       // K = k0 + hi*8 + {0..7}
        bu.h[1] = *(const v8h*)(bp + 16);  // K = k0 + hi*8 + 16 + {0..7}
        v16h bmat = bu.v;

        acc = __builtin_amdgcn_wmma_f32_16x16x32_f16(
            /*neg_a=*/false, a.v, /*neg_b=*/false, bmat,
            /*c_mod=*/(short)0, acc, /*reuse_a=*/false, /*reuse_b=*/false);
    }

    const int col = c0 + lm;        // c = b*SEQLEN + l
    const int bb  = col >> 11;
    const int ll  = col & (SEQLEN - 1);
#pragma unroll
    for (int r = 0; r < 8; ++r) {
        int e = m0 + r + hi * 8;
        out[((size_t)bb * DMODEL + e) * SEQLEN + ll] = acc[r] + bias[e];
    }
}

// ---------------------------------------------------------------------------
extern "C" void kernel_launch(void* const* d_in, const int* in_sizes, int n_in,
                              void* d_out, int out_size, void* d_ws, size_t ws_size,
                              hipStream_t stream) {
    const float* x    = (const float*)d_in[0];   // [8,128,2048]
    const float* A    = (const float*)d_in[1];   // [128,64,64]
    const float* Bm   = (const float*)d_in[2];   // [128,64]
    const float* Cm   = (const float*)d_in[3];   // [128,64]
    const float* Dv   = (const float*)d_in[4];   // [128]
    const float* W    = (const float*)d_in[5];   // [128,128]
    const float* bias = (const float*)d_in[6];   // [128]
    float* out = (float*)d_out;                  // [8,128,2048] f32

    // Workspace: Yt (f16, 16384 x 128 = 4 MB) then Wh (f16, 128x128 = 32 KB)
    _Float16* Yt = (_Float16*)d_ws;
    _Float16* Wh = (_Float16*)((char*)d_ws + (size_t)NCOL * DMODEL * sizeof(_Float16));

    convert_w_kernel<<<(DMODEL * DMODEL + 255) / 256, 256, 0, stream>>>(W, Wh);

    // 1024 (b,d) waves, 8 waves (256 threads) per block
    s4_scan_kernel<<<BATCH * DMODEL / 8, 256, 0, stream>>>(x, A, Bm, Cm, Dv, Yt);

    // 1024 N-tiles x 8 M-tiles; 8 waves per block
    proj_wmma_kernel<<<dim3(NCOL / 16 / 8, DMODEL / 16), 256, 0, stream>>>(Wh, Yt, bias, out);
}
